// CausalSelfAttention_61795989455492
// MI455X (gfx1250) — compile-verified, matrix operand order checked by reading, not run on hardware
//
#include <hip/hip_runtime.h>
#include <cstdint>
#include <cstddef>

// B=4 T=512 C=1024 H=16 D=64  S_PREV=1536 P=256 WINDOW=256  S=2048  BH=64  M=B*T=2048

typedef __attribute__((ext_vector_type(16))) __bf16 v16bf;
typedef __attribute__((ext_vector_type(8)))  float  v8f;
typedef __attribute__((ext_vector_type(4)))  unsigned int v4u;
typedef __attribute__((ext_vector_type(8)))  int v8i;
typedef __attribute__((ext_vector_type(4)))  int v4i;

#if defined(__has_builtin)
#if __has_builtin(__builtin_amdgcn_tensor_load_to_lds) && __has_builtin(__builtin_amdgcn_s_wait_tensorcnt)
#define USE_TDM 1
#endif
#endif
#ifndef USE_TDM
#define USE_TDM 0
#endif

__device__ __forceinline__ v8f wmma_bf(v16bf a, v16bf b, v8f c) {
  // (neg_a, A, neg_b, B, c_mod, C, reuse_a, reuse_b) -> v_wmma_f32_16x16x32_bf16
  return __builtin_amdgcn_wmma_f32_16x16x32_bf16(false, a, false, b, (short)0, c, false, false);
}

#if USE_TDM
// TDM: DMA a 2-D tile (rows x 64 bf16 elements, row stride 64) global -> LDS.
// Descriptor layout per CDNA5 ISA ch.8 (D# groups 0/1; groups 2/3 zero for 2-D).
// This toolchain's builtin is the 6-arg form:
//   (uint32x4 g0, int32x8 g1, int32x4 g2, int32x4 g3, int32x8 spare, i32 cpol)
__device__ __forceinline__ void tdm_load_tile(const __bf16* gsrc, unsigned lds_addr, int rows) {
  unsigned long long ga = (unsigned long long)(uintptr_t)gsrc;
  v4u g0;
  g0[0] = 1u;                                             // count=1, user descriptor
  g0[1] = lds_addr;                                       // LDS byte address (flat addr[31:0])
  g0[2] = (unsigned)(ga & 0xffffffffu);                   // global_addr[31:0]
  g0[3] = (unsigned)((ga >> 32) & 0x01ffffffu) | (2u << 30); // global_addr[56:32] | type=2
  v8i g1;
  g1[0] = (int)(1u << 16);                                // workgroup_mask=0, data_size=1 (2 bytes)
  g1[1] = (int)(64u << 16);                               // tensor_dim0 = 64 (bits 79:48, low part)
  g1[2] = (int)(((unsigned)rows & 0xffffu) << 16);        // tensor_dim1 (bits 111:80, low part)
  g1[3] = (int)(64u << 16);                               // tile_dim0 = 64 (bits 127:112)
  g1[4] = (int)((unsigned)rows & 0xffffu);                // tile_dim1 = rows (bits 143:128)
  g1[5] = 64;                                             // tensor_dim0_stride = 64 (bits 207:160)
  g1[6] = 0;
  g1[7] = 0;
  v4i gz4 = {0, 0, 0, 0};
  v8i gz8 = {0, 0, 0, 0, 0, 0, 0, 0};
  __builtin_amdgcn_tensor_load_to_lds(g0, g1, gz4, gz4, gz8, 0);
}
#endif

// ---------------- elementwise converts ----------------
__global__ __launch_bounds__(256) void k_f32_to_bf16(const float* __restrict__ s,
                                                     __bf16* __restrict__ d, int n) {
  int i = blockIdx.x * 256 + threadIdx.x;
  if (i < n) d[i] = (__bf16)s[i];
}

// cached_k/v (BH,1536,64) f32 -> concat bf16 buffer (BH,2048,64) at s in [0,1536)
__global__ __launch_bounds__(256) void k_copy_cache_kv(const float* __restrict__ s,
                                                       __bf16* __restrict__ d, int n) {
  int i = blockIdx.x * 256 + threadIdx.x;
  if (i >= n) return;
  int dd = i & 63;
  int ss = (i >> 6) % 1536;
  int bh = i / (1536 * 64);
  d[((size_t)bh * 2048 + ss) * 64 + dd] = (__bf16)s[i];
}

// ---------------- RoPE + scatter qkv -> Q / K(concat tail) / V(concat tail) ----------------
__global__ __launch_bounds__(256) void k_rope_scatter(const __bf16* __restrict__ qkv,
                                                      __bf16* __restrict__ Qbf,
                                                      __bf16* __restrict__ Kbf,
                                                      __bf16* __restrict__ Vbf,
                                                      const int* __restrict__ startp) {
  int idx = blockIdx.x * 256 + threadIdx.x;   // B*T*H*32 = 2^20 threads, one per rotary pair
  int j = idx & 31;
  int h = (idx >> 5) & 15;
  int t = (idx >> 9) & 511;
  int b = idx >> 18;
  float pos = (float)(startp[0] + t);
  float inv = __powf(10000.0f, -(float)j * (1.0f / 32.0f));
  float ang = pos * inv;
  float cs = __cosf(ang), sn = __sinf(ang);
  size_t mrow = ((size_t)(b * 512 + t)) * 3072;
  size_t hb = (size_t)h * 64;
  int bh = b * 16 + h;
  float q1 = (float)qkv[mrow + hb + j];
  float q2 = (float)qkv[mrow + hb + 32 + j];
  size_t qo = ((size_t)bh * 512 + t) * 64;
  Qbf[qo + j]      = (__bf16)(q1 * cs - q2 * sn);
  Qbf[qo + 32 + j] = (__bf16)(q1 * sn + q2 * cs);
  float k1 = (float)qkv[mrow + 1024 + hb + j];
  float k2 = (float)qkv[mrow + 1024 + hb + 32 + j];
  size_t ko = ((size_t)bh * 2048 + 1536 + t) * 64;
  Kbf[ko + j]      = (__bf16)(k1 * cs - k2 * sn);
  Kbf[ko + 32 + j] = (__bf16)(k1 * sn + k2 * cs);
  Vbf[ko + j]      = qkv[mrow + 2048 + hb + j];
  Vbf[ko + 32 + j] = qkv[mrow + 2048 + hb + 32 + j];
}

// ---------------- bf16 GEMM:  C[M,N] = A[M,K] * Bw[N,K]^T  ----------------
template <bool F32OUT>
__global__ __launch_bounds__(128) void k_gemm_bf16(const __bf16* __restrict__ A,
                                                   const __bf16* __restrict__ Bw,
                                                   void* __restrict__ Cout, int N, int K) {
  int lane = threadIdx.x & 31;
  int wave = threadIdx.x >> 5;
  int n0 = blockIdx.x * 64;
  int m0 = blockIdx.y * 64 + wave * 16;
  int ln = lane & 15;   // A: row-in-tile / B: col-in-tile / C: col
  int kh = lane >> 4;   // K half selector
  const __bf16* arow = A + (size_t)(m0 + ln) * K;
  v8f acc[4] = {};
  for (int kc = 0; kc < K; kc += 32) {
    __builtin_prefetch(arow + kc + 256, 0, 0);
    v16bf af = *(const v16bf*)(arow + kc + kh * 16);
#pragma unroll
    for (int nb = 0; nb < 4; ++nb) {
      const __bf16* brow = Bw + (size_t)(n0 + nb * 16 + ln) * K + kc + kh * 16;
      v16bf bf = *(const v16bf*)brow;
      acc[nb] = wmma_bf(af, bf, acc[nb]);
    }
  }
#pragma unroll
  for (int nb = 0; nb < 4; ++nb) {
#pragma unroll
    for (int r = 0; r < 8; ++r) {
      int row = m0 + r + 8 * kh;      // C layout: vgpr r -> rows r / r+8
      int col = n0 + nb * 16 + ln;
      float v = acc[nb][r];
      if (F32OUT) ((float*)Cout)[(size_t)row * N + col] = v;
      else        ((__bf16*)Cout)[(size_t)row * N + col] = (__bf16)v;
    }
  }
}

// ---------------- fused streaming attention ----------------
// One wave per (b,h, 16-row q tile).  Online softmax over [prefix(256) | banded S=2048].
__global__ __launch_bounds__(32) void k_attention(const __bf16* __restrict__ Qbf,
                                                  const __bf16* __restrict__ Kbf,
                                                  const __bf16* __restrict__ Vbf,
                                                  const __bf16* __restrict__ CVbf,
                                                  const float* __restrict__ pref,
                                                  __bf16* __restrict__ Ybf) {
  __shared__ alignas(64) __bf16 Ktile[32][64];
  __shared__ alignas(64) __bf16 Vtile[32][64];
  __shared__ alignas(16) float  Stile[16][36];

  const int lane = threadIdx.x;
  const int tile = blockIdx.x;        // 0 .. 64*32-1
  const int qt = tile & 31;
  const int bh = tile >> 5;
  const int qb = qt * 16;
  const int m  = lane & 15;
  const int kh = lane >> 4;

  const __bf16* qrow = Qbf + ((size_t)bh * 512 + qb + m) * 64;
  const v16bf qf0 = *(const v16bf*)(qrow + kh * 16);        // d 0..31
  const v16bf qf1 = *(const v16bf*)(qrow + 32 + kh * 16);   // d 32..63

  v8f o[4] = {};
  float rmax = -1e30f, rsum = 0.0f;

#if !USE_TDM
  auto stage32 = [&](const __bf16* src, __bf16 (*dst)[64]) {
    const uint4* s4 = (const uint4*)(src + (size_t)lane * 64);
    uint4* d4 = (uint4*)&dst[lane][0];
#pragma unroll
    for (int j = 0; j < 8; ++j) d4[j] = s4[j];
  };
#endif

  // logits l[16] held lane=row (m), kh selects 16-col half of the 32-col chunk
  auto softmax_pv = [&](const float* l, const __bf16 (*Vt)[64]) {
    float tmax = l[0];
#pragma unroll
    for (int i = 1; i < 16; ++i) tmax = fmaxf(tmax, l[i]);
    tmax = fmaxf(tmax, __shfl_xor(tmax, 16, 32));
    float nm = fmaxf(rmax, tmax);
    float corr = __expf(rmax - nm);
    rmax = nm;
    v16bf af;
    float lsum = 0.0f;
#pragma unroll
    for (int i = 0; i < 16; ++i) {
      float p = __expf(l[i] - nm);
      lsum += p;
      af[i] = (__bf16)p;
    }
    lsum += __shfl_xor(lsum, 16, 32);
    rsum = rsum * corr + lsum;
    float cs[8];
#pragma unroll
    for (int r = 0; r < 8; ++r) cs[r] = __shfl(corr, r + 8 * kh, 32);
#pragma unroll
    for (int nb = 0; nb < 4; ++nb)
#pragma unroll
      for (int r = 0; r < 8; ++r) o[nb][r] *= cs[r];
#pragma unroll
    for (int nb = 0; nb < 4; ++nb) {
      v16bf bf;
#pragma unroll
      for (int i = 0; i < 16; ++i) bf[i] = Vt[kh * 16 + i][nb * 16 + m];
      o[nb] = wmma_bf(af, bf, o[nb]);
    }
  };

  // ---- Phase 1: prefix (logits streamed from memory, V = cache_v) ----
  for (int pc = 0; pc < 256; pc += 32) {
#if USE_TDM
    tdm_load_tile(CVbf + ((size_t)bh * 256 + pc) * 64,
                  (unsigned)(uintptr_t)&Vtile[0][0], 32);
    __builtin_amdgcn_s_wait_tensorcnt(0);
    __asm__ volatile("" ::: "memory");
#else
    stage32(CVbf + ((size_t)bh * 256 + pc) * 64, Vtile);
#endif
    const float* pr = pref + ((size_t)bh * 512 + qb + m) * 256 + pc + kh * 16;
    float l[16];
#pragma unroll
    for (int i = 0; i < 16; ++i) l[i] = pr[i];
    softmax_pv(l, Vtile);
  }

  // ---- Phase 2: banded attention over S = 4 reps of 512 ----
  int jjlo = qb - 256;
  if (jjlo < 0) jjlo = 0;
  jjlo &= ~31;
  for (int rep = 0; rep < 4; ++rep) {
    for (int jj = jjlo; jj <= qb + 15; jj += 32) {
      int sb = rep * 512 + jj;
#if USE_TDM
      tdm_load_tile(Kbf + ((size_t)bh * 2048 + sb) * 64,
                    (unsigned)(uintptr_t)&Ktile[0][0], 32);
      tdm_load_tile(Vbf + ((size_t)bh * 2048 + sb) * 64,
                    (unsigned)(uintptr_t)&Vtile[0][0], 32);
      __builtin_amdgcn_s_wait_tensorcnt(0);
      __asm__ volatile("" ::: "memory");
#else
      stage32(Kbf + ((size_t)bh * 2048 + sb) * 64, Ktile);
      stage32(Vbf + ((size_t)bh * 2048 + sb) * 64, Vtile);
#endif
#pragma unroll
      for (int t2 = 0; t2 < 2; ++t2) {
        const __bf16* krow = &Ktile[t2 * 16 + m][0];
        v16bf bk0 = *(const v16bf*)(krow + kh * 16);
        v16bf bk1 = *(const v16bf*)(krow + 32 + kh * 16);
        v8f sf = {};
        sf = wmma_bf(qf0, bk0, sf);
        sf = wmma_bf(qf1, bk1, sf);
        // mask + C-layout -> LDS (row-major) so probs can be re-read in A layout
#pragma unroll
        for (int r = 0; r < 8; ++r) {
          int row = qb + r + 8 * kh;
          int col = jj + t2 * 16 + m;
          int dd = row - col;
          Stile[r + 8 * kh][t2 * 16 + m] =
              (dd >= 0 && dd <= 256) ? sf[r] * 0.125f : -1e30f;
        }
      }
      float l[16];
#pragma unroll
      for (int i = 0; i < 16; ++i) l[i] = Stile[m][kh * 16 + i];
      softmax_pv(l, Vtile);
    }
  }

  // ---- Epilogue: normalize, store y as (B,T,H*D) bf16 ----
  float rs[8];
#pragma unroll
  for (int r = 0; r < 8; ++r) rs[r] = 1.0f / __shfl(rsum, r + 8 * kh, 32);
  int b = bh >> 4, h = bh & 15;
#pragma unroll
  for (int nb = 0; nb < 4; ++nb)
#pragma unroll
    for (int r = 0; r < 8; ++r) {
      int row = qb + r + 8 * kh;
      int col = h * 64 + nb * 16 + m;
      Ybf[((size_t)(b * 512 + row)) * 1024 + col] = (__bf16)(o[nb][r] * rs[r]);
    }
}

// ---------------- host launch ----------------
extern "C" void kernel_launch(void* const* d_in, const int* in_sizes, int n_in,
                              void* d_out, int out_size, void* d_ws, size_t ws_size,
                              hipStream_t stream) {
  (void)in_sizes; (void)n_in; (void)out_size; (void)ws_size;
  const float* x        = (const float*)d_in[0];
  const float* attn_w   = (const float*)d_in[1];
  const float* proj_w   = (const float*)d_in[2];
  const float* cached_k = (const float*)d_in[3];
  const float* cached_v = (const float*)d_in[4];
  const float* att_pref = (const float*)d_in[5];
  const float* cache_v  = (const float*)d_in[6];
  const int*   start_i  = (const int*)d_in[7];

  char* ws = (char*)d_ws;
  size_t off = 0;
  auto take = [&](size_t bytes) {
    void* p = ws + off;
    off += (bytes + 255) & ~(size_t)255;
    return p;
  };
  __bf16* XBF   = (__bf16*)take((size_t)2048 * 1024 * 2);  // x bf16 (M,K)
  __bf16* WATT  = (__bf16*)take((size_t)3072 * 1024 * 2);  // c_attn_w bf16 (N,K)
  __bf16* WPROJ = (__bf16*)take((size_t)1024 * 1024 * 2);  // c_proj_w bf16 (N,K)
  __bf16* QKV   = (__bf16*)take((size_t)2048 * 3072 * 2);  // qkv bf16 (M,3C)
  __bf16* QBF   = (__bf16*)take((size_t)64 * 512 * 64 * 2);   // (BH,T,D)
  __bf16* KBF   = (__bf16*)take((size_t)64 * 2048 * 64 * 2);  // concat (BH,S,D)
  __bf16* VBF   = (__bf16*)take((size_t)64 * 2048 * 64 * 2);  // concat (BH,S,D)
  __bf16* CVBF  = (__bf16*)take((size_t)64 * 256 * 64 * 2);   // (BH,P,D)
  __bf16* YBF   = (__bf16*)take((size_t)2048 * 1024 * 2);     // y bf16 (M,C)

  int n;
  n = 2048 * 1024;  k_f32_to_bf16<<<(n + 255) / 256, 256, 0, stream>>>(x, XBF, n);
  n = 3072 * 1024;  k_f32_to_bf16<<<(n + 255) / 256, 256, 0, stream>>>(attn_w, WATT, n);
  n = 1024 * 1024;  k_f32_to_bf16<<<(n + 255) / 256, 256, 0, stream>>>(proj_w, WPROJ, n);
  n = 64 * 256 * 64; k_f32_to_bf16<<<(n + 255) / 256, 256, 0, stream>>>(cache_v, CVBF, n);
  n = 64 * 1536 * 64;
  k_copy_cache_kv<<<(n + 255) / 256, 256, 0, stream>>>(cached_k, KBF, n);
  k_copy_cache_kv<<<(n + 255) / 256, 256, 0, stream>>>(cached_v, VBF, n);

  // qkv = x @ c_attn_w.T   (M=2048, N=3072, K=1024)
  k_gemm_bf16<false><<<dim3(3072 / 64, 2048 / 64), 128, 0, stream>>>(XBF, WATT, (void*)QKV, 3072, 1024);

  n = 4 * 512 * 16 * 32;
  k_rope_scatter<<<(n + 255) / 256, 256, 0, stream>>>(QKV, QBF, KBF, VBF, start_i);

  k_attention<<<64 * 32, 32, 0, stream>>>(QBF, KBF, VBF, CVBF, att_pref, YBF);

  // out = y @ c_proj_w.T   (M=2048, N=1024, K=1024), f32 output
  k_gemm_bf16<true><<<dim3(1024 / 64, 2048 / 64), 128, 0, stream>>>(YBF, WPROJ, d_out, 1024, 1024);
}
